// AdaptiveBlock_81432579932596
// MI455X (gfx1250) — matrix operand order repaired and checked by reading, not compiled, not run on hardware
//
#include <hip/hip_runtime.h>

// ---------------------------------------------------------------------------
// Transformer block (B=4,S=1024,D=1024,H=16,HD=64, rel-pos attention), gfx1250.
// All GEMM math: v_wmma_f32_16x16x32_bf16 (fp32 accum). Data movement:
// global_load_async_to_lds_b128 (ASYNCcnt) + TDM tensor_load_to_lds (TENSORcnt).
// ---------------------------------------------------------------------------

#define DEV __device__ __forceinline__

static constexpr int BB    = 4;
static constexpr int SEQ   = 1024;
static constexpr int DIM   = 1024;
static constexpr int NH    = 16;
static constexpr int HD    = 64;
static constexpr int MROWS = BB * SEQ;   // 4096
static constexpr int RPAD  = 272;        // 257 rel positions padded to 17*16

typedef __attribute__((ext_vector_type(16))) __bf16 bf16x16;
typedef __attribute__((ext_vector_type(8)))  float  f32x8;

union Frag {                 // one WMMA A/B operand: 16 bf16 per lane
  uint4   q[2];
  bf16x16 v;
};

DEV unsigned short f2bf(float f) {          // round-to-nearest-even f32->bf16
  unsigned u = __float_as_uint(f);
  u += 0x7FFFu + ((u >> 16) & 1u);
  return (unsigned short)(u >> 16);
}

DEV f32x8 wmma_bf16(const Frag& a, const Frag& b, f32x8 c) {
  return __builtin_amdgcn_wmma_f32_16x16x32_bf16(
      false, a.v, false, b.v, (short)0, c, false, false);
}

// ---- CDNA5 async copy global->LDS (ASYNCcnt) --------------------------------
DEV void async_b128(unsigned lds_off, const unsigned short* g) {
  asm volatile("global_load_async_to_lds_b128 %0, %1, off"
               :: "v"(lds_off), "v"(g) : "memory");
}
DEV void wait_async0() { asm volatile("s_wait_asynccnt 0x0" ::: "memory"); }
DEV void wait_async4() { asm volatile("s_wait_asynccnt 0x4" ::: "memory"); }

// ---- CDNA5 Tensor Data Mover (TENSORcnt) ------------------------------------
#if __has_builtin(__builtin_amdgcn_tensor_load_to_lds)
#define HAVE_TDM 1
#else
#define HAVE_TDM 0
#endif

#if HAVE_TDM
typedef __attribute__((ext_vector_type(4))) unsigned u32x4;
typedef __attribute__((ext_vector_type(8))) int      i32x8;
typedef __attribute__((ext_vector_type(4))) int      i32x4;

// 2D tile load: tileH rows x tileW bf16 elements, row stride rowStrideElems.
// D# built per CDNA5 ISA (count=1, type=2, data_size=2B).
DEV void tdm_load_2d(unsigned ldsOff, const void* gptr,
                     unsigned tileW, unsigned tileH, unsigned rowStrideElems) {
  unsigned long long ga = (unsigned long long)(size_t)gptr;
  u32x4 g0 = { 1u,                                      // count=1, user mode
               ldsOff,                                  // lds_addr
               (unsigned)ga,                            // global_addr[31:0]
               ((unsigned)((ga >> 32) & 0x01FFFFFFu)) | (2u << 30) }; // type=2
  i32x8 g1 = { (int)(1u << 16),                  // data_size=1 (2 bytes)
               (int)(tileW << 16),               // tensor_dim0[15:0]
               (int)(tileH << 16),               // tensor_dim1[15:0]
               (int)(tileW << 16),               // tile_dim0
               (int)tileH,                       // tile_dim1 (tile_dim2=0)
               (int)rowStrideElems,              // tensor_dim0_stride[31:0]
               0, 0 };
  i32x4 z4 = {0, 0, 0, 0};
#if defined(__clang_major__) && (__clang_major__ >= 23)
  i32x8 z8 = {0, 0, 0, 0, 0, 0, 0, 0};
  __builtin_amdgcn_tensor_load_to_lds(g0, g1, z4, z4, z8, 0);
#else
  __builtin_amdgcn_tensor_load_to_lds(g0, g1, z4, z4, 0);
#endif
}
#endif

// ---------------------------------------------------------------------------
// fp32 [K][N] -> bf16 transposed [N][K]  (weights, done once; LDS-tiled)
// grid = (N/32, K/32), 256 threads
// ---------------------------------------------------------------------------
__global__ void __launch_bounds__(256)
cvt_transpose_kernel(const float* __restrict__ in,
                     unsigned short* __restrict__ out, int K, int N) {
  __shared__ unsigned short t[32][33];
  const int tx = threadIdx.x & 31, ty = threadIdx.x >> 5;
  const int n0 = blockIdx.x * 32, k0 = blockIdx.y * 32;
#pragma unroll
  for (int i = 0; i < 4; i++) {
    int k = k0 + ty + i * 8;
    t[tx][ty + i * 8] = f2bf(in[(size_t)k * N + n0 + tx]);
  }
  __syncthreads();
#pragma unroll
  for (int i = 0; i < 4; i++) {
    int n = n0 + ty + i * 8;
    out[(size_t)n * K + k0 + tx] = t[ty + i * 8][tx];
  }
}

// bf16 V [B*S][D] -> Vt [B*H][HD][S]  (per-head transpose)
// grid = (S/32, HD/32, B*H), 256 threads
__global__ void __launch_bounds__(256)
transpose_v_kernel(const unsigned short* __restrict__ V,
                   unsigned short* __restrict__ Vt) {
  __shared__ unsigned short t[32][33];
  const int tx = threadIdx.x & 31, ty = threadIdx.x >> 5;
  const int bh = blockIdx.z, b = bh >> 4, h = bh & 15;
  const int s0 = blockIdx.x * 32, d0 = blockIdx.y * 32;
#pragma unroll
  for (int i = 0; i < 4; i++) {
    int s = s0 + ty + i * 8;
    t[tx][ty + i * 8] = V[(size_t)(b * SEQ + s) * DIM + h * HD + d0 + tx];
  }
  __syncthreads();
#pragma unroll
  for (int i = 0; i < 4; i++) {
    int d = d0 + ty + i * 8;
    Vt[((size_t)bh * HD + d) * SEQ + s0 + tx] = t[ty + i * 8][tx];
  }
}

// rel_embed [257][64] -> bf16 padded [272][64]
__global__ void relpad_kernel(const float* __restrict__ rel,
                              unsigned short* __restrict__ out) {
  int i = blockIdx.x * 256 + threadIdx.x;
  if (i < RPAD * HD) {
    int r = i / HD;
    out[i] = (r < 257) ? f2bf(rel[i]) : (unsigned short)0;
  }
}

// ---------------------------------------------------------------------------
// LayerNorm over D=1024, one 256-thread block per row, bf16 output
// ---------------------------------------------------------------------------
__global__ void __launch_bounds__(256)
layernorm_bf16_kernel(const float* __restrict__ X, const float* __restrict__ g,
                      const float* __restrict__ bta,
                      unsigned short* __restrict__ out) {
  __shared__ float r1[256], r2[256];
  const float* row = X + (size_t)blockIdx.x * DIM;
  float s = 0.f, ss = 0.f;
  for (int i = threadIdx.x; i < DIM; i += 256) {
    float v = row[i]; s += v; ss += v * v;
  }
  r1[threadIdx.x] = s; r2[threadIdx.x] = ss;
  __syncthreads();
  for (int st = 128; st > 0; st >>= 1) {
    if (threadIdx.x < st) {
      r1[threadIdx.x] += r1[threadIdx.x + st];
      r2[threadIdx.x] += r2[threadIdx.x + st];
    }
    __syncthreads();
  }
  float mu   = r1[0] * (1.0f / DIM);
  float var  = r2[0] * (1.0f / DIM) - mu * mu;
  float rstd = rsqrtf(var + 1e-5f);
  unsigned short* orow = out + (size_t)blockIdx.x * DIM;
  for (int i = threadIdx.x; i < DIM; i += 256)
    orow[i] = f2bf((row[i] - mu) * rstd * g[i] + bta[i]);
}

// ---------------------------------------------------------------------------
// WMMA GEMM: C[M,N] = A[M,K](bf16) @ Bw[N,K](bf16, pre-transposed)
//            + bias (+resid)(+gelu).
// Block tile 128x128, 256 threads (8 waves as 4Mx2N); wave = 32x64 sub-tile
// (2 A-frags x 4 B-frags = 8 WMMA per k-step). LDS double-buffered with
// compile-time buffer indices (k-loop unrolled x2); all staging via
// global_load_async_to_lds_b128 so next tile's DMA overlaps current WMMAs.
// grid = (N/128, M/128)
// ---------------------------------------------------------------------------
__global__ void __launch_bounds__(256)
gemm_bf16_kernel(const unsigned short* __restrict__ A,
                 const unsigned short* __restrict__ Bw,
                 const float* __restrict__ bias,
                 const float* __restrict__ resid,
                 float* __restrict__ outF, unsigned short* __restrict__ outB,
                 int Mdim, int Ndim, int Kdim, int do_gelu) {
  (void)Mdim;
  __shared__ unsigned short As[2][128][32];   // [m][k] 16KB
  __shared__ unsigned short Bs[2][128][32];   // [n][k] 16KB

  const int tid  = threadIdx.x;
  const int wave = tid >> 5, lane = tid & 31;
  const int hf = lane >> 4, lm = lane & 15;
  const int m0 = blockIdx.y * 128, n0 = blockIdx.x * 128;
  const int mw = (wave >> 1) * 32;            // wave row base within tile
  const int nw = (wave & 1) * 64;             // wave col base within tile

  // async chunk assignment: A,B = 512 x 16B chunks each; 4 chunks/thread
  const int ca0 = tid, ca1 = tid + 256;
  const unsigned short* gA0 = A  + (size_t)(m0 + (ca0 >> 2)) * Kdim + (ca0 & 3) * 8;
  const unsigned short* gA1 = A  + (size_t)(m0 + (ca1 >> 2)) * Kdim + (ca1 & 3) * 8;
  const unsigned short* gB0 = Bw + (size_t)(n0 + (ca0 >> 2)) * Kdim + (ca0 & 3) * 8;
  const unsigned short* gB1 = Bw + (size_t)(n0 + (ca1 >> 2)) * Kdim + (ca1 & 3) * 8;
  unsigned lA0[2], lA1[2], lB0[2], lB1[2];
#pragma unroll
  for (int bf = 0; bf < 2; bf++) {
    lA0[bf] = (unsigned)(size_t)&As[bf][ca0 >> 2][(ca0 & 3) * 8];
    lA1[bf] = (unsigned)(size_t)&As[bf][ca1 >> 2][(ca1 & 3) * 8];
    lB0[bf] = (unsigned)(size_t)&Bs[bf][ca0 >> 2][(ca0 & 3) * 8];
    lB1[bf] = (unsigned)(size_t)&Bs[bf][ca1 >> 2][(ca1 & 3) * 8];
  }

  f32x8 acc0[4] = {{}, {}, {}, {}};   // rows mw..mw+15
  f32x8 acc1[4] = {{}, {}, {}, {}};   // rows mw+16..mw+31

  auto issue = [&](int buf) {
    async_b128(lA0[buf], gA0); async_b128(lA1[buf], gA1);
    async_b128(lB0[buf], gB0); async_b128(lB1[buf], gB1);
    gA0 += 32; gA1 += 32; gB0 += 32; gB1 += 32;
  };
  auto compute = [&](int buf) {
    Frag a0, a1;
    const uint4* ap0 = (const uint4*)&As[buf][mw + lm][hf * 16];
    a0.q[0] = ap0[0]; a0.q[1] = ap0[1];
    const uint4* ap1 = (const uint4*)&As[buf][mw + 16 + lm][hf * 16];
    a1.q[0] = ap1[0]; a1.q[1] = ap1[1];
#pragma unroll
    for (int nt = 0; nt < 4; nt++) {
      Frag b;
      const uint4* bp = (const uint4*)&Bs[buf][nw + nt * 16 + lm][hf * 16];
      b.q[0] = bp[0]; b.q[1] = bp[1];
      acc0[nt] = wmma_bf16(a0, b, acc0[nt]);
      acc1[nt] = wmma_bf16(a1, b, acc1[nt]);
    }
  };

  const int nK = Kdim >> 5;   // always even (K = 1024 or 4096)
  issue(0);                   // tile 0 in flight
  for (int kt = 0; kt < nK; kt += 2) {
    issue(1);                 // tile kt+1 in flight during compute of kt
    wait_async4();            // tile kt landed (async copies finish in order)
    __syncthreads();
    compute(0);
    __syncthreads();          // everyone done reading buf0
    if (kt + 2 < nK) { issue(0); wait_async4(); }
    else             { wait_async0(); }
    __syncthreads();          // tile kt+1 landed
    compute(1);
    __syncthreads();          // everyone done reading buf1
  }

#pragma unroll
  for (int nt = 0; nt < 4; nt++) {
    int col = n0 + nw + nt * 16 + lm;
    float bv = bias ? bias[col] : 0.f;
#pragma unroll
    for (int j = 0; j < 8; j++) {
      int row0 = m0 + mw + j + hf * 8;
      float v0 = acc0[nt][j] + bv;
      float v1 = acc1[nt][j] + bv;
      if (resid) {
        v0 += resid[(size_t)row0 * Ndim + col];
        v1 += resid[(size_t)(row0 + 16) * Ndim + col];
      }
      if (do_gelu) {
        v0 = 0.5f * v0 * (1.0f + erff(v0 * 0.70710678118654752f));
        v1 = 0.5f * v1 * (1.0f + erff(v1 * 0.70710678118654752f));
      }
      if (outF) {
        outF[(size_t)row0 * Ndim + col]        = v0;
        outF[(size_t)(row0 + 16) * Ndim + col] = v1;
      } else {
        outB[(size_t)row0 * Ndim + col]        = f2bf(v0);
        outB[(size_t)(row0 + 16) * Ndim + col] = f2bf(v1);
      }
    }
  }
}

// ---------------------------------------------------------------------------
// Qrel[bh][q][r] = (1/8) * sum_d Q[b,q,h,d] * relpad[r][d]
// grid = (SEQ/64, 17, B*H), 128 threads
// ---------------------------------------------------------------------------
__global__ void __launch_bounds__(128)
qrel_kernel(const unsigned short* __restrict__ Qb,
            const unsigned short* __restrict__ relpad,
            float* __restrict__ qrel) {
  const int wave = threadIdx.x >> 5, lane = threadIdx.x & 31;
  const int hf = lane >> 4, lm = lane & 15;
  const int bh = blockIdx.z, b = bh >> 4, h = bh & 15;
  const int q0 = blockIdx.x * 64 + wave * 16;
  const int r0 = blockIdx.y * 16;

  const unsigned short* qrow = Qb + ((size_t)(b * SEQ + q0 + lm) * DIM) + h * HD;
  Frag a0, a1;
  a0.q[0] = *(const uint4*)(qrow + hf * 16);
  a0.q[1] = *(const uint4*)(qrow + hf * 16 + 8);
  a1.q[0] = *(const uint4*)(qrow + 32 + hf * 16);
  a1.q[1] = *(const uint4*)(qrow + 32 + hf * 16 + 8);

  const unsigned short* rrow = relpad + (size_t)(r0 + lm) * HD;
  Frag b0, b1;
  b0.q[0] = *(const uint4*)(rrow + hf * 16);
  b0.q[1] = *(const uint4*)(rrow + hf * 16 + 8);
  b1.q[0] = *(const uint4*)(rrow + 32 + hf * 16);
  b1.q[1] = *(const uint4*)(rrow + 32 + hf * 16 + 8);

  f32x8 acc = {};
  acc = wmma_bf16(a0, b0, acc);
  acc = wmma_bf16(a1, b1, acc);

#pragma unroll
  for (int j = 0; j < 8; j++) {
    int q = q0 + j + hf * 8;
    qrel[((size_t)bh * SEQ + q) * RPAD + r0 + lm] = acc[j] * 0.125f;
  }
}

// ---------------------------------------------------------------------------
// Flash attention with relative bias. grid = (SEQ/64, B*H), 128 threads.
// K tile staged by TDM (TENSORcnt), V tile (pre-transposed Vt) by async copies.
// ---------------------------------------------------------------------------
__global__ void __launch_bounds__(128)
attn_kernel(const unsigned short* __restrict__ Qb,
            const unsigned short* __restrict__ Kb,
            const unsigned short* __restrict__ Vt,
            const float* __restrict__ qrel,
            unsigned short* __restrict__ Ob) {
  __shared__ unsigned short Kls[32][64];       // [k][d] 4KB (TDM destination)
  __shared__ unsigned short VT[64][32];        // [d][k] 4KB
  __shared__ unsigned short Pls[4][16][32];    // per-wave P scratch 4KB

  const int tid = threadIdx.x;
  const int wave = tid >> 5, lane = tid & 31;
  const int hf = lane >> 4, lm = lane & 15;
  const int bh = blockIdx.y, b = bh >> 4, h = bh & 15;
  const int q0 = blockIdx.x * 64 + wave * 16;

  // Q fragments (held in registers for the whole kernel)
  const unsigned short* qrow = Qb + ((size_t)(b * SEQ + q0 + lm) * DIM) + h * HD;
  Frag qa0, qa1;
  qa0.q[0] = *(const uint4*)(qrow + hf * 16);
  qa0.q[1] = *(const uint4*)(qrow + hf * 16 + 8);
  qa1.q[0] = *(const uint4*)(qrow + 32 + hf * 16);
  qa1.q[1] = *(const uint4*)(qrow + 32 + hf * 16 + 8);

  // per-thread async source pointers for the V tile (2 x 16B chunks/thread)
  const int cv0 = tid, cv1 = tid + 128;        // 256 chunks: d=c>>2, ch=c&3
  const unsigned short* gV0 =
      Vt + ((size_t)bh * HD + (cv0 >> 2)) * SEQ + (cv0 & 3) * 8;
  const unsigned short* gV1 =
      Vt + ((size_t)bh * HD + (cv1 >> 2)) * SEQ + (cv1 & 3) * 8;
  const unsigned lV0 = (unsigned)(size_t)&VT[cv0 >> 2][(cv0 & 3) * 8];
  const unsigned lV1 = (unsigned)(size_t)&VT[cv1 >> 2][(cv1 & 3) * 8];
#if !HAVE_TDM
  const int ck0 = tid, ck1 = tid + 128;        // 256 chunks: k=c>>3, ch=c&7
  const unsigned short* gK0 =
      Kb + ((size_t)(b * SEQ + (ck0 >> 3)) * DIM) + h * HD + (ck0 & 7) * 8;
  const unsigned short* gK1 =
      Kb + ((size_t)(b * SEQ + (ck1 >> 3)) * DIM) + h * HD + (ck1 & 7) * 8;
  const unsigned lK0 = (unsigned)(size_t)&Kls[ck0 >> 3][(ck0 & 7) * 8];
  const unsigned lK1 = (unsigned)(size_t)&Kls[ck1 >> 3][(ck1 & 7) * 8];
#endif

  f32x8 o0 = {}, o1 = {}, o2 = {}, o3 = {};
  float mrow[8], lrow[8];
#pragma unroll
  for (int j = 0; j < 8; j++) { mrow[j] = -1e30f; lrow[j] = 0.f; }

  const float* qr = qrel + (size_t)bh * SEQ * RPAD;

  for (int k0 = 0; k0 < SEQ; k0 += 32) {
    __syncthreads();                       // prior iter done reading Kls/VT
#if HAVE_TDM
    if (tid < 32)                          // wave0 drives the Tensor Data Mover
      tdm_load_2d((unsigned)(size_t)&Kls[0][0],
                  Kb + ((size_t)(b * SEQ + k0) * DIM) + h * HD,
                  /*tileW=*/HD, /*tileH=*/32, /*rowStrideElems=*/DIM);
#else
    async_b128(lK0, gK0 + (size_t)k0 * DIM);
    async_b128(lK1, gK1 + (size_t)k0 * DIM);
#endif
    async_b128(lV0, gV0 + k0);             // V tile: contiguous rows from Vt
    async_b128(lV1, gV1 + k0);
    wait_async0();
#if HAVE_TDM
    if (tid < 32) __builtin_amdgcn_s_wait_tensorcnt(0);
#endif
    __syncthreads();

    // ---- scores: S = Q Kt, two 16-col tiles, K-dim 64 = 2 wmma each ----
    f32x8 s0 = {}, s1 = {};
    {
      Frag kb;
      const uint4* kp = (const uint4*)&Kls[0 * 16 + lm][hf * 16];
      kb.q[0] = kp[0]; kb.q[1] = kp[1];
      s0 = wmma_bf16(qa0, kb, s0);
      kp = (const uint4*)&Kls[0 * 16 + lm][32 + hf * 16];
      kb.q[0] = kp[0]; kb.q[1] = kp[1];
      s0 = wmma_bf16(qa1, kb, s0);
    }
    {
      Frag kb;
      const uint4* kp = (const uint4*)&Kls[1 * 16 + lm][hf * 16];
      kb.q[0] = kp[0]; kb.q[1] = kp[1];
      s1 = wmma_bf16(qa0, kb, s1);
      kp = (const uint4*)&Kls[1 * 16 + lm][32 + hf * 16];
      kb.q[0] = kp[0]; kb.q[1] = kp[1];
      s1 = wmma_bf16(qa1, kb, s1);
    }

    // ---- scale + relative bias (Qrel gather, L2 resident) ----
    float p0[8], p1[8];
#pragma unroll
    for (int j = 0; j < 8; j++) {
      int q  = q0 + j + hf * 8;
      int kA = k0 + lm, kB = k0 + 16 + lm;
      int rA = min(max(kA - q, -128), 128) + 128;
      int rB = min(max(kB - q, -128), 128) + 128;
      p0[j] = s0[j] * 0.125f + qr[(size_t)q * RPAD + rA];
      p1[j] = s1[j] * 0.125f + qr[(size_t)q * RPAD + rB];
    }

    // ---- online softmax (row reductions across 16-lane halves) ----
#pragma unroll
    for (int j = 0; j < 8; j++) {
      float mx = fmaxf(p0[j], p1[j]);
      for (int m = 8; m >= 1; m >>= 1) mx = fmaxf(mx, __shfl_xor(mx, m));
      float mn = fmaxf(mrow[j], mx);
      float alpha = __expf(mrow[j] - mn);
      mrow[j] = mn;
      p0[j] = __expf(p0[j] - mn);
      p1[j] = __expf(p1[j] - mn);
      float rs = p0[j] + p1[j];
      for (int m = 8; m >= 1; m >>= 1) rs += __shfl_xor(rs, m);
      lrow[j] = lrow[j] * alpha + rs;
      o0[j] *= alpha; o1[j] *= alpha; o2[j] *= alpha; o3[j] *= alpha;
    }

    // ---- P -> bf16, wave-local LDS transpose into A layout ----
#pragma unroll
    for (int j = 0; j < 8; j++) {
      Pls[wave][j + hf * 8][lm]      = f2bf(p0[j]);
      Pls[wave][j + hf * 8][16 + lm] = f2bf(p1[j]);
    }
    Frag pa;
    {
      const uint4* pp = (const uint4*)&Pls[wave][lm][hf * 16];
      pa.q[0] = pp[0];
      pa.q[1] = pp[1];
    }

    // ---- O += P(16x32) @ V(32x64): 4 d-tiles ----
    {
      Frag vb; const uint4* vp = (const uint4*)&VT[0 * 16 + lm][hf * 16];
      vb.q[0] = vp[0]; vb.q[1] = vp[1];
      o0 = wmma_bf16(pa, vb, o0);
    }
    {
      Frag vb; const uint4* vp = (const uint4*)&VT[1 * 16 + lm][hf * 16];
      vb.q[0] = vp[0]; vb.q[1] = vp[1];
      o1 = wmma_bf16(pa, vb, o1);
    }
    {
      Frag vb; const uint4* vp = (const uint4*)&VT[2 * 16 + lm][hf * 16];
      vb.q[0] = vp[0]; vb.q[1] = vp[1];
      o2 = wmma_bf16(pa, vb, o2);
    }
    {
      Frag vb; const uint4* vp = (const uint4*)&VT[3 * 16 + lm][hf * 16];
      vb.q[0] = vp[0]; vb.q[1] = vp[1];
      o3 = wmma_bf16(pa, vb, o3);
    }
  }

  // ---- normalize and store bf16 [B*S, D] ----
#pragma unroll
  for (int j = 0; j < 8; j++) {
    float inv = 1.0f / lrow[j];
    int q = q0 + j + hf * 8;
    unsigned short* op = Ob + ((size_t)(b * SEQ + q) * DIM) + h * HD;
    op[0 * 16 + lm] = f2bf(o0[j] * inv);
    op[1 * 16 + lm] = f2bf(o1[j] * inv);
    op[2 * 16 + lm] = f2bf(o2[j] * inv);
    op[3 * 16 + lm] = f2bf(o3[j] * inv);
  }
}

// ---------------------------------------------------------------------------
// Host orchestration
// ---------------------------------------------------------------------------
extern "C" void kernel_launch(void* const* d_in, const int* in_sizes, int n_in,
                              void* d_out, int out_size, void* d_ws,
                              size_t ws_size, hipStream_t stream) {
  (void)in_sizes; (void)n_in; (void)out_size; (void)ws_size;
  const float* x    = (const float*)d_in[0];
  const float* wq   = (const float*)d_in[1];
  const float* bq   = (const float*)d_in[2];
  const float* wk   = (const float*)d_in[3];
  const float* bk   = (const float*)d_in[4];
  const float* wv   = (const float*)d_in[5];
  const float* bv   = (const float*)d_in[6];
  const float* wo   = (const float*)d_in[7];
  const float* bo   = (const float*)d_in[8];
  const float* rel  = (const float*)d_in[9];
  const float* ln1g = (const float*)d_in[10];
  const float* ln1b = (const float*)d_in[11];
  const float* ln2g = (const float*)d_in[12];
  const float* ln2b = (const float*)d_in[13];
  const float* w1   = (const float*)d_in[14];
  const float* b1   = (const float*)d_in[15];
  const float* w2   = (const float*)d_in[16];
  const float* b2   = (const float*)d_in[17];
  float* out = (float*)d_out;

  char* p = (char*)d_ws;
  auto carve = [&](size_t bytes) -> void* {
    void* r = (void*)p;
    p += (bytes + 255) & ~(size_t)255;
    return r;
  };
  const size_t actBF = (size_t)MROWS * DIM * sizeof(unsigned short);
  unsigned short* hb   = (unsigned short*)carve(actBF);
  unsigned short* qb   = (unsigned short*)carve(actBF);
  unsigned short* kb   = (unsigned short*)carve(actBF);
  unsigned short* vb_  = (unsigned short*)carve(actBF);
  unsigned short* vtb  = (unsigned short*)carve(actBF);                // Vt
  unsigned short* aob  = (unsigned short*)carve(actBF);
  float*          x2   = (float*)carve((size_t)MROWS * DIM * 4);
  unsigned short* h2   = (unsigned short*)carve(actBF);
  unsigned short* ffm  = (unsigned short*)carve((size_t)MROWS * 4 * DIM * 2);
  unsigned short* wqT  = (unsigned short*)carve((size_t)DIM * DIM * 2);
  unsigned short* wkT  = (unsigned short*)carve((size_t)DIM * DIM * 2);
  unsigned short* wvT  = (unsigned short*)carve((size_t)DIM * DIM * 2);
  unsigned short* woT  = (unsigned short*)carve((size_t)DIM * DIM * 2);
  unsigned short* w1T  = (unsigned short*)carve((size_t)DIM * 4 * DIM * 2);
  unsigned short* w2T  = (unsigned short*)carve((size_t)4 * DIM * DIM * 2);
  unsigned short* relp = (unsigned short*)carve((size_t)RPAD * HD * 2);
  float*          qrl  = (float*)carve((size_t)BB * NH * SEQ * RPAD * 4);

  // weights -> bf16, transposed to [N][K]
  dim3 gT(DIM / 32, DIM / 32);
  cvt_transpose_kernel<<<gT, 256, 0, stream>>>(wq, wqT, DIM, DIM);
  cvt_transpose_kernel<<<gT, 256, 0, stream>>>(wk, wkT, DIM, DIM);
  cvt_transpose_kernel<<<gT, 256, 0, stream>>>(wv, wvT, DIM, DIM);
  cvt_transpose_kernel<<<gT, 256, 0, stream>>>(wo, woT, DIM, DIM);
  cvt_transpose_kernel<<<dim3(4 * DIM / 32, DIM / 32), 256, 0, stream>>>(
      w1, w1T, DIM, 4 * DIM);
  cvt_transpose_kernel<<<dim3(DIM / 32, 4 * DIM / 32), 256, 0, stream>>>(
      w2, w2T, 4 * DIM, DIM);
  relpad_kernel<<<(RPAD * HD + 255) / 256, 256, 0, stream>>>(rel, relp);

  // LN1 -> h (bf16)
  layernorm_bf16_kernel<<<MROWS, 256, 0, stream>>>(x, ln1g, ln1b, hb);

  // Q,K,V projections
  dim3 gQKV(DIM / 128, MROWS / 128);
  gemm_bf16_kernel<<<gQKV, 256, 0, stream>>>(hb, wqT, bq, nullptr, nullptr, qb,
                                             MROWS, DIM, DIM, 0);
  gemm_bf16_kernel<<<gQKV, 256, 0, stream>>>(hb, wkT, bk, nullptr, nullptr, kb,
                                             MROWS, DIM, DIM, 0);
  gemm_bf16_kernel<<<gQKV, 256, 0, stream>>>(hb, wvT, bv, nullptr, nullptr, vb_,
                                             MROWS, DIM, DIM, 0);

  // per-head V transpose -> Vt [B*H][HD][S]
  transpose_v_kernel<<<dim3(SEQ / 32, HD / 32, BB * NH), 256, 0, stream>>>(vb_,
                                                                           vtb);

  // Qrel = scale * Q @ rel^T
  qrel_kernel<<<dim3(SEQ / 64, RPAD / 16, BB * NH), 128, 0, stream>>>(qb, relp,
                                                                      qrl);

  // flash attention with rel bias
  attn_kernel<<<dim3(SEQ / 64, BB * NH), 128, 0, stream>>>(qb, kb, vtb, qrl,
                                                           aob);

  // out-projection + residual -> x2
  gemm_bf16_kernel<<<gQKV, 256, 0, stream>>>(aob, woT, bo, x, x2, nullptr,
                                             MROWS, DIM, DIM, 0);

  // LN2 -> h2
  layernorm_bf16_kernel<<<MROWS, 256, 0, stream>>>(x2, ln2g, ln2b, h2);

  // FFN1 (exact GELU) -> ffm
  gemm_bf16_kernel<<<dim3(4 * DIM / 128, MROWS / 128), 256, 0, stream>>>(
      h2, w1T, b1, nullptr, nullptr, ffm, MROWS, 4 * DIM, DIM, 1);

  // FFN2 + residual -> out
  gemm_bf16_kernel<<<dim3(DIM / 128, MROWS / 128), 256, 0, stream>>>(
      ffm, w2T, b2, x2, out, nullptr, MROWS, DIM, 4 * DIM, 0);
}
